// InfoNCELoss_58342835749209
// MI455X (gfx1250) — matrix operand order repaired and checked by reading, not compiled, not run on hardware
//
#include <hip/hip_runtime.h>

typedef __attribute__((ext_vector_type(16))) _Float16 v16h;
typedef __attribute__((ext_vector_type(8)))  _Float16 v8h;
typedef __attribute__((ext_vector_type(8)))  float    v8f;
typedef __attribute__((ext_vector_type(4)))  unsigned int v4u;
typedef __attribute__((ext_vector_type(8)))  int      v8i;
typedef __attribute__((ext_vector_type(4)))  int      v4i;

#define NEG_INF (-1e9f)
#define INV_T   (1.0f / 0.07f)

static constexpr int BB = 64;    // batch
static constexpr int LL = 256;   // tokens
static constexpr int HH = 64;    // head dim
static constexpr int QK_HALFS = LL * HH;                     // 16384 halves per batch row
static constexpr size_t TENSOR_HALFS = (size_t)BB * LL * HH; // 1,048,576 halves per tensor

// ---------------------------------------------------------------------------
// Kernel 1: convert f32 embeddings to f16 for the WMMA path.
// ---------------------------------------------------------------------------
__global__ void cvt_f16_kernel(const float* __restrict__ q,
                               const float* __restrict__ k,
                               _Float16* __restrict__ qh,
                               _Float16* __restrict__ kh,
                               int n) {
    int i = blockIdx.x * blockDim.x + threadIdx.x;
    if (i < n) {
        qh[i] = (_Float16)q[i];
        kh[i] = (_Float16)k[i];
    }
}

// ---------------------------------------------------------------------------
// Tensor Data Mover: DMA a contiguous 32 KB (16384 x f16) block from global
// memory into LDS. Descriptor per CDNA5 ISA §8 (D# groups 0/1; 1-D tile).
// Issued once by wave 0; completion via s_wait_tensorcnt, published by barrier.
// ---------------------------------------------------------------------------
#if __has_builtin(__builtin_amdgcn_tensor_load_to_lds)
#define HAVE_TDM 1
__device__ __forceinline__ void tdm_copy_32k_to_lds(unsigned lds_addr,
                                                    unsigned long long gaddr) {
    v4u g0;
    g0.x = 1u;                                       // count=1 (valid user D#)
    g0.y = lds_addr;                                 // LDS byte address
    g0.z = (unsigned)(gaddr & 0xFFFFFFFFull);        // global_addr[31:0]
    g0.w = (unsigned)((gaddr >> 32) & 0x1FFFFFFull)  // global_addr[56:32]
         | (2u << 30);                               // type=2 ("image")
    v8i g1;
    g1[0] = (int)(1u << 16);       // data_size=1 (2 bytes/elem); no multicast/pad
    g1[1] = (int)(16384u << 16);   // tensor_dim0[15:0] = 16384 in bits [63:48]
    g1[2] = 0;                     // tensor_dim0 hi = 0, tensor_dim1 = 0
    g1[3] = (int)(16384u << 16);   // tile_dim0 = 16384 in bits [127:112]
    g1[4] = 0;                     // tile_dim1 = 0 (unused), tile_dim2 = 0
    g1[5] = 16384;                 // tensor_dim0_stride lo
    g1[6] = 0;
    g1[7] = 0;
    v4i z4 = {0, 0, 0, 0};
#if defined(__clang_major__) && __clang_major__ >= 23
    v8i z8 = {0, 0, 0, 0, 0, 0, 0, 0};
    __builtin_amdgcn_tensor_load_to_lds(g0, g1, z4, z4, z8, 0);
#else
    __builtin_amdgcn_tensor_load_to_lds(g0, g1, z4, z4, 0);
#endif
}
#else
#define HAVE_TDM 0
#endif

// ---------------------------------------------------------------------------
// Kernel 2: per (b,c) pair, compute ColBERT score with WMMA f16 GEMM tiles.
// grid = (64, 64): blockIdx.x = c (key batch), blockIdx.y = b (query batch).
// 256 threads = 8 waves; each wave owns M-tiles {wave, wave+8}.
// ---------------------------------------------------------------------------
__global__ void __launch_bounds__(256)
colbert_scores_kernel(const _Float16* __restrict__ qh,
                      const _Float16* __restrict__ kh,
                      const int* __restrict__ q_mask,
                      const int* __restrict__ k_mask,
                      float* __restrict__ scores) {
    __shared__ __align__(16) _Float16 sK[QK_HALFS];   // 32 KB: K_c tile
    __shared__ float s_part[16];                      // per (wave, half-group) partials

    const int c   = blockIdx.x;
    const int b   = blockIdx.y;
    const int tid = threadIdx.x;

    const int wave   = tid >> 5;
    const int lane   = tid & 31;
    const int l15    = lane & 15;
    const int hiHalf = lane >> 4;   // 0 for lanes 0-15, 1 for lanes 16-31

    // ---- Stage K_c into LDS ------------------------------------------------
#if HAVE_TDM
    if (wave == 0) {
        tdm_copy_32k_to_lds((unsigned)(size_t)&sK[0],
                            (unsigned long long)(size_t)(kh + (size_t)c * QK_HALFS));
        __builtin_amdgcn_s_wait_tensorcnt(0);
    }
#else
    {
        const uint4* gk  = (const uint4*)(kh + (size_t)c * QK_HALFS);
        uint4*       sk4 = (uint4*)sK;
        #pragma unroll
        for (int i = 0; i < 8; ++i)
            sk4[tid + i * 256] = gk[tid + i * 256];
    }
#endif

    // Per-lane key-mask bits: bit nt = k_mask[c][nt*16 + l15] (this lane's
    // column in N-tile nt). Computed once; inner loop is then VMEM-free.
    unsigned kbits = 0;
    #pragma unroll
    for (int nt = 0; nt < 16; ++nt)
        kbits |= (k_mask[c * LL + nt * 16 + l15] != 0 ? 1u : 0u) << nt;

    __syncthreads();

    float partial = 0.0f;

    // Two M-tiles per wave (16 tiles of 16 query rows).
    for (int mi = 0; mi < 2; ++mi) {
        const int mt = wave + mi * 8;

        // A fragment (16x32 f16, ISA layout): lanes 0-15 hold K {0..7,16..23},
        // lanes 16-31 hold K {8..15,24..31}, row M = lane&15. Direct from global
        // (Q is tiny and L2-resident; only 32 KB per block total).
        const _Float16* aptr = qh + (size_t)b * QK_HALFS + (mt * 16 + l15) * HH;
        const int alo = hiHalf ? 8 : 0;
        v8h a0lo = *(const v8h*)(aptr + 0  + alo);
        v8h a0hi = *(const v8h*)(aptr + 16 + alo);
        v8h a1lo = *(const v8h*)(aptr + 32 + alo);
        v8h a1hi = *(const v8h*)(aptr + 48 + alo);
        v16h a0 = __builtin_shufflevector(a0lo, a0hi, 0,1,2,3,4,5,6,7,8,9,10,11,12,13,14,15);
        v16h a1 = __builtin_shufflevector(a1lo, a1hi, 0,1,2,3,4,5,6,7,8,9,10,11,12,13,14,15);

        float rmax[8];
        #pragma unroll
        for (int r = 0; r < 8; ++r) rmax[r] = NEG_INF;

        // Sweep all 16 N-tiles of key tokens.
        for (int nt = 0; nt < 16; ++nt) {
            // B fragment (32x16 f16): lane group n = lane&15 is the key token
            // (column), lanes 0-15 carry K {0..7,16..23}, lanes 16-31 carry K {8..15,24..31}.
            const _Float16* bptr = sK + (nt * 16 + l15) * HH + hiHalf * 8;
            v8h b0lo = *(const v8h*)(bptr + 0);
            v8h b0hi = *(const v8h*)(bptr + 16);
            v8h b1lo = *(const v8h*)(bptr + 32);
            v8h b1hi = *(const v8h*)(bptr + 48);
            v16h bf0 = __builtin_shufflevector(b0lo, b0hi, 0,1,2,3,4,5,6,7,8,9,10,11,12,13,14,15);
            v16h bf1 = __builtin_shufflevector(b1lo, b1hi, 0,1,2,3,4,5,6,7,8,9,10,11,12,13,14,15);

            v8f acc = {};
            acc = __builtin_amdgcn_wmma_f32_16x16x32_f16(false, a0, false, bf0,
                                                         (short)0, acc, false, false);
            acc = __builtin_amdgcn_wmma_f32_16x16x32_f16(false, a1, false, bf1,
                                                         (short)0, acc, false, false);

            // Column mask for this lane's column (same for all 8 rows).
            const bool kvalid = ((kbits >> nt) & 1u) != 0u;
            #pragma unroll
            for (int r = 0; r < 8; ++r) {
                float v = kvalid ? acc[r] : NEG_INF;
                rmax[r] = fmaxf(rmax[r], v);
            }
        }

        // Reduce the running max across each 16-lane half-group (wave32 xor stays in-group).
        #pragma unroll
        for (int r = 0; r < 8; ++r) {
            float v = rmax[r];
            v = fmaxf(v, __shfl_xor(v, 8, 32));
            v = fmaxf(v, __shfl_xor(v, 4, 32));
            v = fmaxf(v, __shfl_xor(v, 2, 32));
            v = fmaxf(v, __shfl_xor(v, 1, 32));
            rmax[r] = v;
        }

        // C/D layout: VGPR r holds row M = r (lanes 0-15) or M = 8+r (lanes 16-31).
        if (l15 == 0) {
            const int rowbase = mt * 16 + hiHalf * 8;
            #pragma unroll
            for (int r = 0; r < 8; ++r) {
                if (q_mask[b * LL + rowbase + r]) partial += rmax[r];
            }
        }
    }

    // Deterministic reduction of the 16 (wave, half-group) partials.
    if (l15 == 0) s_part[wave * 2 + hiHalf] = partial;
    __syncthreads();
    if (tid == 0) {
        float s = 0.0f;
        #pragma unroll
        for (int i = 0; i < 16; ++i) s += s_part[i];
        scores[b * BB + c] = s * INV_T;
    }
}

// ---------------------------------------------------------------------------
// Kernel 3: row-wise log-softmax, diagonal, label-weighted mean -> scalar loss.
// One block, 64 threads (one row each).
// ---------------------------------------------------------------------------
__global__ void finalize_kernel(const float* __restrict__ scores,
                                const long long* __restrict__ labels,
                                float* __restrict__ out) {
    __shared__ float swd[64];
    __shared__ float sw[64];
    const int t = threadIdx.x;

    float m = NEG_INF;
    for (int j = 0; j < BB; ++j) m = fmaxf(m, scores[t * BB + j]);
    float se = 0.0f;
    for (int j = 0; j < BB; ++j) se += __expf(scores[t * BB + j] - m);
    const float diag = scores[t * BB + t] - m - __logf(se);
    const float w = (float)labels[t];

    swd[t] = w * diag;
    sw[t]  = w;
    __syncthreads();
    for (int s = 32; s > 0; s >>= 1) {
        if (t < s) { swd[t] += swd[t + s]; sw[t] += sw[t + s]; }
        __syncthreads();
    }
    if (t == 0) out[0] = -swd[0] / fmaxf(sw[0], 1.0f);
}

// ---------------------------------------------------------------------------
extern "C" void kernel_launch(void* const* d_in, const int* in_sizes, int n_in,
                              void* d_out, int out_size, void* d_ws, size_t ws_size,
                              hipStream_t stream) {
    const float*     q      = (const float*)d_in[0];
    const float*     k      = (const float*)d_in[1];
    const long long* labels = (const long long*)d_in[2];
    const int*       qm     = (const int*)d_in[3];
    const int*       km     = (const int*)d_in[4];

    char* ws = (char*)d_ws;
    _Float16* qh     = (_Float16*)(ws);                         // 2 MB
    _Float16* kh     = (_Float16*)(ws + TENSOR_HALFS * 2);      // 2 MB
    float*    scores = (float*)(ws + TENSOR_HALFS * 4);         // 16 KB

    const int n = (int)TENSOR_HALFS;  // 1,048,576 elements per tensor
    cvt_f16_kernel<<<n / 256, 256, 0, stream>>>(q, k, qh, kh, n);

    dim3 grid(BB, BB);  // (c, b)
    colbert_scores_kernel<<<grid, 256, 0, stream>>>(qh, kh, qm, km, scores);

    finalize_kernel<<<1, 64, 0, stream>>>(scores, labels, (float*)d_out);
}
